// BiMambaBlock_84138409329238
// MI455X (gfx1250) — matrix operand order repaired and checked by reading, not compiled, not run on hardware
//
#include <hip/hip_runtime.h>
#include <hip/hip_bf16.h>

typedef __bf16 bf16_t;
typedef __attribute__((ext_vector_type(16))) __bf16 v16bf;
typedef __attribute__((ext_vector_type(8)))  float  v8f;

#define BATCH   8
#define SEQL    4096
#define MROWS   (BATCH * SEQL)      // 32768
#define DMODEL  256
#define DINNER  512
#define DSTATE  16
#define NSEG    32
#define SLEN    128                 // NSEG*SLEN == SEQL

union Frag16 { v16bf v; uint4 q[2]; };

__device__ __forceinline__ float sigmoid_f(float x) { return 1.0f / (1.0f + __expf(-x)); }

__device__ __forceinline__ int flip_row(int r, int flip) {
    if (!flip) return r;
    int b = r >> 12;                 // / SEQL
    int l = r & (SEQL - 1);
    return (b << 12) + (SEQL - 1 - l);
}

__device__ __forceinline__ void load_afrag(Frag16& f, const char* base, size_t kb) {
    f.q[0] = *(const uint4*)(base + kb);
    f.q[1] = *(const uint4*)(base + kb + 32);
}
__device__ __forceinline__ void load_bfrag(Frag16& f, const char* base, size_t kb) {
    f.q[0] = *(const uint4*)(base + kb);
    f.q[1] = *(const uint4*)(base + kb + 32);
}

// ---------------------------------------------------------------------------
// f32 -> bf16 conversion
// ---------------------------------------------------------------------------
__global__ void cvt_bf16_kernel(const float* __restrict__ in, bf16_t* __restrict__ out, size_t n) {
    size_t i = (size_t)blockIdx.x * blockDim.x + threadIdx.x;
    if (i < n) out[i] = (bf16_t)in[i];
}

// ---------------------------------------------------------------------------
// bf16 WMMA GEMM, 32x32 per wave (2x2 register-blocked WMMA tiles).
// out[M,N] = A[M,K] * W[N,K]^T (+bias) (+gelu).
// Block: 256 threads = 8 waves as 4(M) x 2(N) -> 128x64 block tile.
// Requires M%128==0, N%64==0, K%32==0.
// ---------------------------------------------------------------------------
template <int ACT>  // 0 = none, 1 = exact GELU
__global__ void __launch_bounds__(256)
wmma_gemm2x2_kernel(const bf16_t* __restrict__ A, const bf16_t* __restrict__ W,
                    const float* __restrict__ bias,
                    float* __restrict__ outF, bf16_t* __restrict__ outB,
                    int N, int K, int flip) {
    const int lane = threadIdx.x & 31;
    const int wave = threadIdx.x >> 5;
    const int sub  = lane & 15;
    const int half = lane >> 4;
    const int m0   = blockIdx.y * 128 + (wave & 3) * 32;
    const int n0   = blockIdx.x * 64 + (wave >> 2) * 32;

    const int ar0 = flip_row(m0 + sub, flip);
    const int ar1 = flip_row(m0 + 16 + sub, flip);
    const char* a0 = (const char*)A + ((size_t)ar0 * K + half * 8) * sizeof(bf16_t);
    const char* a1 = (const char*)A + ((size_t)ar1 * K + half * 8) * sizeof(bf16_t);
    const char* b0 = (const char*)W + ((size_t)(n0 + sub) * K + half * 8) * sizeof(bf16_t);
    const char* b1 = (const char*)W + ((size_t)(n0 + 16 + sub) * K + half * 8) * sizeof(bf16_t);

    v8f c00 = {0.f,0.f,0.f,0.f,0.f,0.f,0.f,0.f};
    v8f c01 = c00, c10 = c00, c11 = c00;

    for (int k = 0; k < K; k += 32) {
        const size_t kb = (size_t)k * sizeof(bf16_t);
        Frag16 fa0, fa1, fb0, fb1;
        load_afrag(fa0, a0, kb);
        load_afrag(fa1, a1, kb);
        load_bfrag(fb0, b0, kb);
        load_bfrag(fb1, b1, kb);
        c00 = __builtin_amdgcn_wmma_f32_16x16x32_bf16(false, fa0.v, false, fb0.v, (short)0, c00, false, false);
        c01 = __builtin_amdgcn_wmma_f32_16x16x32_bf16(false, fa0.v, false, fb1.v, (short)0, c01, false, false);
        c10 = __builtin_amdgcn_wmma_f32_16x16x32_bf16(false, fa1.v, false, fb0.v, (short)0, c10, false, false);
        c11 = __builtin_amdgcn_wmma_f32_16x16x32_bf16(false, fa1.v, false, fb1.v, (short)0, c11, false, false);
    }

    // Epilogue: tile (ti,tj): rows m0+16*ti+8*half+i, col n0+16*tj+sub
    v8f* accs[4] = { &c00, &c01, &c10, &c11 };
#pragma unroll
    for (int t = 0; t < 4; ++t) {
        const int ti = t >> 1, tj = t & 1;
        const int rbase = m0 + 16 * ti + 8 * half;
        const int n = n0 + 16 * tj + sub;
        float bv = bias ? bias[n] : 0.0f;
#pragma unroll
        for (int i = 0; i < 8; ++i) {
            float v = (*accs[t])[i] + bv;
            if (ACT == 1) v = 0.5f * v * (1.0f + erff(v * 0.70710678118f));
            size_t oi = (size_t)(rbase + i) * N + n;
            if (outF) outF[oi] = v;
            if (outB) outB[oi] = (bf16_t)v;
        }
    }
}

// ---------------------------------------------------------------------------
// bf16 WMMA GEMM, 16x16 per wave (for N=48). Requires N%16==0, M%128==0.
// ---------------------------------------------------------------------------
__global__ void __launch_bounds__(256)
wmma_gemm_kernel(const bf16_t* __restrict__ A, const bf16_t* __restrict__ W,
                 float* __restrict__ outF, int N, int K) {
    const int lane = threadIdx.x & 31;
    const int wave = threadIdx.x >> 5;
    const int sub  = lane & 15;
    const int half = lane >> 4;
    const int n0   = blockIdx.x * 16;
    const int m0   = blockIdx.y * 128 + wave * 16;

    const char* aBase = (const char*)A + ((size_t)(m0 + sub) * K + half * 8) * sizeof(bf16_t);
    const char* bBase = (const char*)W + ((size_t)(n0 + sub) * K + half * 8) * sizeof(bf16_t);

    v8f acc = {0.f,0.f,0.f,0.f,0.f,0.f,0.f,0.f};
    for (int k = 0; k < K; k += 32) {
        const size_t kb = (size_t)k * sizeof(bf16_t);
        Frag16 fa, fb;
        load_afrag(fa, aBase, kb);
        load_bfrag(fb, bBase, kb);
        acc = __builtin_amdgcn_wmma_f32_16x16x32_bf16(false, fa.v, false, fb.v, (short)0, acc, false, false);
    }
    const int orow = m0 + 8 * half;
    const int n = n0 + sub;
#pragma unroll
    for (int i = 0; i < 8; ++i)
        outF[(size_t)(orow + i) * N + n] = acc[i];
}

// ---------------------------------------------------------------------------
// Depthwise causal conv (k=4) + bias + SiLU on xi half of xz; write f32 + bf16
// ---------------------------------------------------------------------------
__global__ void conv_silu_kernel(const float* __restrict__ xz,
                                 const float* __restrict__ conv_w,
                                 const float* __restrict__ conv_b,
                                 float* __restrict__ uF, bf16_t* __restrict__ uB) {
    size_t idx = (size_t)blockIdx.x * blockDim.x + threadIdx.x;
    if (idx >= (size_t)MROWS * DINNER) return;
    int c = (int)(idx & (DINNER - 1));
    int m = (int)(idx >> 9);
    int l = m & (SEQL - 1);
    float acc = conv_b[c];
#pragma unroll
    for (int k = 0; k < 4; ++k) {
        int ls = l - 3 + k;
        if (ls >= 0) acc += xz[(size_t)(m - 3 + k) * 1024 + c] * conv_w[c * 4 + k];
    }
    float s = acc * sigmoid_f(acc);
    uF[idx] = s;
    uB[idx] = (bf16_t)s;
}

// ---------------------------------------------------------------------------
// dt = softplus(dbc[:, :16] @ dt_proj_w^T + dt_proj_b)   (K=16, done in VALU)
// ---------------------------------------------------------------------------
__global__ void dt_softplus_kernel(const float* __restrict__ dbc,
                                   const float* __restrict__ dtw,
                                   const float* __restrict__ dtb,
                                   float* __restrict__ dtOut) {
    size_t idx = (size_t)blockIdx.x * blockDim.x + threadIdx.x;
    if (idx >= (size_t)MROWS * DINNER) return;
    int d = (int)(idx & (DINNER - 1));
    size_t m = idx >> 9;
    const float* row = dbc + m * 48;
    const float* wr  = dtw + d * 16;
    float acc = dtb[d];
#pragma unroll
    for (int r = 0; r < 16; ++r) acc += row[r] * wr[r];
    dtOut[idx] = (acc > 20.0f) ? acc : log1pf(__expf(acc));
}

// ---------------------------------------------------------------------------
// Chunked selective scan. Segment-local pass (template: WRITE_Y).
// Pass 0: h0 = 0, record h_end and sum(dt). Pass 1: start from h0, emit y.
// grid = (BATCH, NSEG), block = 512 (one lane per inner channel d)
// ---------------------------------------------------------------------------
template <int WRITE_Y>
__global__ void __launch_bounds__(512)
scan_seg_kernel(const float* dtv, const float* __restrict__ dbc, const float* __restrict__ u,
                const float* __restrict__ A_log, const float* __restrict__ h0,
                float* __restrict__ hend, float* __restrict__ dtsumO, float* ys) {
    __shared__ float sB[SLEN][DSTATE];
    __shared__ float sC[SLEN][DSTATE];
    const int b = blockIdx.x, seg = blockIdx.y, d = threadIdx.x;
    const int l0 = seg * SLEN;

    float Aneg[DSTATE], h[DSTATE];
#pragma unroll
    for (int n = 0; n < DSTATE; ++n) {
        Aneg[n] = -__expf(A_log[d * DSTATE + n]);
        h[n] = WRITE_Y ? h0[((size_t)(b * NSEG + seg) * DINNER + d) * DSTATE + n] : 0.0f;
    }
    for (int i = threadIdx.x; i < SLEN * DSTATE; i += 512) {
        int ll = i >> 4, n = i & 15;
        size_t rb = (size_t)(b * SEQL + l0 + ll) * 48;
        sB[ll][n] = dbc[rb + 16 + n];
        if (WRITE_Y) sC[ll][n] = dbc[rb + 32 + n];
    }
    __syncthreads();

    float dts = 0.0f;
    for (int ll = 0; ll < SLEN; ++ll) {
        size_t idx = (size_t)(b * SEQL + l0 + ll) * DINNER + d;
        float dt = dtv[idx];
        float du = dt * u[idx];
        dts += dt;
        float y = 0.0f;
#pragma unroll
        for (int n = 0; n < DSTATE; ++n) {
            float dA = __expf(dt * Aneg[n]);
            h[n] = dA * h[n] + du * sB[ll][n];
            if (WRITE_Y) y += h[n] * sC[ll][n];
        }
        if (WRITE_Y) ys[idx] = y;   // aliases dtv buffer; element already consumed
    }
    if (!WRITE_Y) {
#pragma unroll
        for (int n = 0; n < DSTATE; ++n)
            hend[((size_t)(b * NSEG + seg) * DINNER + d) * DSTATE + n] = h[n];
        dtsumO[(size_t)(b * NSEG + seg) * DINNER + d] = dts;
    }
}

// Cross-segment carry: diagonal transition product = exp(A * sum(dt))
__global__ void scan_combine_kernel(const float* __restrict__ hend,
                                    const float* __restrict__ dtsum,
                                    const float* __restrict__ A_log,
                                    float* __restrict__ h0) {
    int idx = blockIdx.x * blockDim.x + threadIdx.x;   // (b,d,n): 8*512*16 = 65536
    if (idx >= BATCH * DINNER * DSTATE) return;
    int n = idx & 15;
    int d = (idx >> 4) & (DINNER - 1);
    int b = idx >> 13;
    float Aneg = -__expf(A_log[d * DSTATE + n]);
    float h = 0.0f;
    for (int s = 0; s < NSEG; ++s) {
        size_t sd = (size_t)(b * NSEG + s) * DINNER + d;
        h0[sd * DSTATE + n] = h;
        float P = __expf(Aneg * dtsum[sd]);
        h = P * h + hend[sd * DSTATE + n];
    }
}

// ---------------------------------------------------------------------------
// y = (ys + D*u) * silu(z);  write bf16 for out_proj
// ---------------------------------------------------------------------------
__global__ void ygate_kernel(const float* __restrict__ ys, const float* __restrict__ u,
                             const float* __restrict__ Dp, const float* __restrict__ xz,
                             bf16_t* __restrict__ outB) {
    size_t idx = (size_t)blockIdx.x * blockDim.x + threadIdx.x;
    if (idx >= (size_t)MROWS * DINNER) return;
    int d = (int)(idx & (DINNER - 1));
    size_t m = idx >> 9;
    float y = ys[idx] + Dp[d] * u[idx];
    float z = xz[m * 1024 + DINNER + d];
    outB[idx] = (bf16_t)(y * z * sigmoid_f(z));
}

// ---------------------------------------------------------------------------
// LayerNorm(a + bmat[maybe-flipped rows]); wave-per-row (wave32), 8 rows/block.
// Optional f32/bf16 outputs; optional scaled accumulate into final output.
// ---------------------------------------------------------------------------
__global__ void __launch_bounds__(256)
add_ln_kernel(const float* __restrict__ a, const float* __restrict__ bm,
              const float* __restrict__ w, const float* __restrict__ bias,
              float* __restrict__ outF, bf16_t* __restrict__ outB,
              float* accOut, float accScale, int accAdd, int flip) {
    const int lane = threadIdx.x & 31;
    const int m = blockIdx.x * 8 + (threadIdx.x >> 5);
    const int bmr = flip_row(m, flip);
    float v[8], s = 0.0f, s2 = 0.0f;
#pragma unroll
    for (int j = 0; j < 8; ++j) {
        int c = lane + 32 * j;
        float t = a[(size_t)m * DMODEL + c] + bm[(size_t)bmr * DMODEL + c];
        v[j] = t; s += t; s2 += t * t;
    }
#pragma unroll
    for (int off = 16; off > 0; off >>= 1) {
        s  += __shfl_xor(s, off, 32);
        s2 += __shfl_xor(s2, off, 32);
    }
    float mu = s * (1.0f / DMODEL);
    float rstd = rsqrtf(s2 * (1.0f / DMODEL) - mu * mu + 1e-5f);
#pragma unroll
    for (int j = 0; j < 8; ++j) {
        int c = lane + 32 * j;
        float o = (v[j] - mu) * rstd * w[c] + bias[c];
        size_t oi = (size_t)m * DMODEL + c;
        if (outF) outF[oi] = o;
        if (outB) outB[oi] = (bf16_t)o;
        if (accOut) { float r = o * accScale; accOut[oi] = accAdd ? (accOut[oi] + r) : r; }
    }
}

// ---------------------------------------------------------------------------
extern "C" void kernel_launch(void* const* d_in, const int* in_sizes, int n_in,
                              void* d_out, int out_size, void* d_ws, size_t ws_size,
                              hipStream_t stream) {
    (void)in_sizes; (void)n_in; (void)out_size; (void)ws_size;
    const float* x = (const float*)d_in[0];

    // ---- workspace layout ----
    char* ws = (char*)d_ws;
    size_t off = 0;
    auto alloc = [&](size_t bytes) -> char* {
        char* p = ws + off;
        off = (off + bytes + 255) & ~(size_t)255;
        return p;
    };
    bf16_t* x_bf  = (bf16_t*)alloc((size_t)MROWS * DMODEL * 2);
    bf16_t* w_in  = (bf16_t*)alloc((size_t)1024 * 256 * 2);
    bf16_t* w_xp  = (bf16_t*)alloc((size_t)48 * 512 * 2);
    bf16_t* w_out = (bf16_t*)alloc((size_t)256 * 512 * 2);
    bf16_t* w_f1  = (bf16_t*)alloc((size_t)1024 * 256 * 2);
    bf16_t* w_f2  = (bf16_t*)alloc((size_t)256 * 1024 * 2);
    float*  xz    = (float*) alloc((size_t)MROWS * 1024 * 4);
    float*  uF    = (float*) alloc((size_t)MROWS * DINNER * 4);
    bf16_t* actB  = (bf16_t*)alloc((size_t)MROWS * 1024 * 2);   // xi_bf16 -> ygate -> ffn1 act
    float*  dbc   = (float*) alloc((size_t)MROWS * 48 * 4);
    float*  dtb   = (float*) alloc((size_t)MROWS * DINNER * 4); // dt, then overwritten by ys
    float*  mo    = (float*) alloc((size_t)MROWS * DMODEL * 4); // mamba out, then ffn2 out
    float*  ylnF  = (float*) alloc((size_t)MROWS * DMODEL * 4);
    bf16_t* ylnB  = (bf16_t*)alloc((size_t)MROWS * DMODEL * 2);
    float*  hend  = (float*) alloc((size_t)BATCH * NSEG * DINNER * DSTATE * 4);
    float*  h0b   = (float*) alloc((size_t)BATCH * NSEG * DINNER * DSTATE * 4);
    float*  dtsum = (float*) alloc((size_t)BATCH * NSEG * DINNER * 4);

    // x -> bf16 (shared by both branches)
    {
        size_t n = (size_t)MROWS * DMODEL;
        cvt_bf16_kernel<<<(unsigned)((n + 255) / 256), 256, 0, stream>>>(x, x_bf, n);
    }

    const size_t ne = (size_t)MROWS * DINNER;
    const unsigned gElem = (unsigned)((ne + 255) / 256);
    const dim3 gN1024(1024 / 64, MROWS / 128), gN256(256 / 64, MROWS / 128), gN48(48 / 16, MROWS / 128);
    const dim3 gScan(BATCH, NSEG);

    for (int br = 0; br < 2; ++br) {
        const int base = 1 + br * 17;
        const int flip = br;  // branch 1 is the reversed branch
        const float* in_proj_w = (const float*)d_in[base + 0];
        const float* conv_w    = (const float*)d_in[base + 1];
        const float* conv_b    = (const float*)d_in[base + 2];
        const float* x_proj_w  = (const float*)d_in[base + 3];
        const float* dt_proj_w = (const float*)d_in[base + 4];
        const float* dt_proj_b = (const float*)d_in[base + 5];
        const float* A_log     = (const float*)d_in[base + 6];
        const float* Dp        = (const float*)d_in[base + 7];
        const float* out_projw = (const float*)d_in[base + 8];
        const float* ln1_w     = (const float*)d_in[base + 9];
        const float* ln1_b     = (const float*)d_in[base + 10];
        const float* ln2_w     = (const float*)d_in[base + 11];
        const float* ln2_b     = (const float*)d_in[base + 12];
        const float* ffn_w1    = (const float*)d_in[base + 13];
        const float* ffn_b1    = (const float*)d_in[base + 14];
        const float* ffn_w2    = (const float*)d_in[base + 15];
        const float* ffn_b2    = (const float*)d_in[base + 16];

        // weights -> bf16
        cvt_bf16_kernel<<<(1024 * 256 + 255) / 256, 256, 0, stream>>>(in_proj_w, w_in, 1024 * 256);
        cvt_bf16_kernel<<<(48 * 512 + 255) / 256, 256, 0, stream>>>(x_proj_w, w_xp, 48 * 512);
        cvt_bf16_kernel<<<(256 * 512 + 255) / 256, 256, 0, stream>>>(out_projw, w_out, 256 * 512);
        cvt_bf16_kernel<<<(1024 * 256 + 255) / 256, 256, 0, stream>>>(ffn_w1, w_f1, 1024 * 256);
        cvt_bf16_kernel<<<(256 * 1024 + 255) / 256, 256, 0, stream>>>(ffn_w2, w_f2, 256 * 1024);

        // in_proj: xz = x(flipped?) @ W^T        [M,1024]
        wmma_gemm2x2_kernel<0><<<gN1024, 256, 0, stream>>>(x_bf, w_in, nullptr, xz, nullptr, 1024, 256, flip);
        // conv + silu -> u (f32 + bf16)
        conv_silu_kernel<<<gElem, 256, 0, stream>>>(xz, conv_w, conv_b, uF, actB);
        // x_proj: dbc = u @ W^T                  [M,48]
        wmma_gemm_kernel<<<gN48, 256, 0, stream>>>(actB, w_xp, dbc, 48, 512);
        // dt = softplus(...)
        dt_softplus_kernel<<<gElem, 256, 0, stream>>>(dbc, dt_proj_w, dt_proj_b, dtb);
        // chunked selective scan (ys overwrites dtb)
        scan_seg_kernel<0><<<gScan, 512, 0, stream>>>(dtb, dbc, uF, A_log, nullptr, hend, dtsum, nullptr);
        scan_combine_kernel<<<(BATCH * DINNER * DSTATE + 255) / 256, 256, 0, stream>>>(hend, dtsum, A_log, h0b);
        scan_seg_kernel<1><<<gScan, 512, 0, stream>>>(dtb, dbc, uF, A_log, h0b, nullptr, nullptr, dtb);
        // gate
        ygate_kernel<<<gElem, 256, 0, stream>>>(dtb, uF, Dp, xz, actB);
        // out_proj -> mo [M,256]
        wmma_gemm2x2_kernel<0><<<gN256, 256, 0, stream>>>(actB, w_out, nullptr, mo, nullptr, 256, 512, 0);
        // ln1: LN(x + unflip(mo)) -> yln (f32 + bf16)
        add_ln_kernel<<<MROWS / 8, 256, 0, stream>>>(x, mo, ln1_w, ln1_b, ylnF, ylnB,
                                                     nullptr, 0.0f, 0, flip);
        // ffn1 (+bias, exact gelu) -> actB [M,1024] bf16
        wmma_gemm2x2_kernel<1><<<gN1024, 256, 0, stream>>>(ylnB, w_f1, ffn_b1, nullptr, actB, 1024, 256, 0);
        // ffn2 (+bias) -> mo [M,256] f32
        wmma_gemm2x2_kernel<0><<<gN256, 256, 0, stream>>>(actB, w_f2, ffn_b2, mo, nullptr, 256, 1024, 0);
        // ln2: LN(yln + ffn2); accumulate 0.5 * result into d_out
        add_ln_kernel<<<MROWS / 8, 256, 0, stream>>>(ylnF, mo, ln2_w, ln2_b, nullptr, nullptr,
                                                     (float*)d_out, 0.5f, br, 0);
    }
}
